// _ObjectAttentionBlock_79534204387442
// MI455X (gfx1250) — matrix operand order repaired and checked by reading, not compiled, not run on hardware
//
#include <hip/hip_runtime.h>
#include <hip/hip_bf16.h>
#include <stdint.h>

// ---------------------------------------------------------------------------
// ObjectAttentionBlock fused for gfx1250 (MI455X, wave32, WMMA bf16)
// B=8, Cin=512, H=W=128 (HW=16384), Kc=256, Kp=150 (padded to 160)
//
// LDS B-operand layout is fragment-native: for each 32-wide K block, column n
// holds 16 contiguous dwords (dword jj = bf16 pair (2jj, 2jj+1)), with a
// 20-dword column stride (16B-aligned, bank-conflict-free for lanes 0-15).
// A B fragment is then two ds_load_b128 per lane, straight into the WMMA
// operand registers -- no register shuffling.
// ---------------------------------------------------------------------------

typedef __attribute__((ext_vector_type(16))) __bf16    v16bf;
typedef __attribute__((ext_vector_type(8)))  float     v8f;
typedef __attribute__((ext_vector_type(4)))  uint32_t  v4u;

#define CS 20                 // column stride in dwords within a K-block
#define KB 1280               // dwords per 32-K block (64 cols * CS)

union FragU { v16bf v; uint32_t u[8]; v4u q[2]; };

__device__ __forceinline__ uint16_t f2bf(float f) {
  uint32_t u = __float_as_uint(f);
  uint32_t r = (u + 0x7FFFu + ((u >> 16) & 1u)) >> 16;   // round-to-nearest-even
  return (uint16_t)r;
}
__device__ __forceinline__ uint32_t pack2bf(float lo, float hi) {
  return (uint32_t)f2bf(lo) | ((uint32_t)f2bf(hi) << 16);
}

// A fragment (16x32 bf16 tile, row-major source, ld in elements).
// ISA: lanes 0-15 row M=lane, VGPR0..3 = K 0..7, VGPR4..7 = K 16..23;
// lanes 16-31 same rows, K 8..15 / 24..31  -> two contiguous 16B loads/lane.
__device__ __forceinline__ v16bf load_a(const uint16_t* A, int ld, int m0, int k0, int lane) {
  int r  = lane & 15;
  int kh = (lane >> 4) << 3;                    // 0 or 8
  const uint32_t* p = (const uint32_t*)(A + (size_t)(m0 + r) * ld + (k0 + kh));
  FragU f;
  f.q[0] = *(const v4u*)p;                      // K = k0+kh .. +7
  f.q[1] = *(const v4u*)(p + 8);                // K = k0+kh+16 .. +23
  return f.v;
}

// B fragment (32x16 bf16 tile) from fragment-native LDS layout.
// ISA: lanes 0-15 col N=lane hold K 0..7, 16..23 (pairs per VGPR); lanes 16-31 K 8..15, 24..31.
__device__ __forceinline__ v16bf load_b(const uint32_t* B, int k0, int n0, int lane) {
  int n = n0 + (lane & 15);
  const uint32_t* p = B + (k0 >> 5) * KB + n * CS + ((lane >> 4) << 2);
  FragU f;
  f.q[0] = *(const v4u*)p;                      // 4 dwords = K 8h..8h+7
  f.q[1] = *(const v4u*)(p + 8);                // 4 dwords = K 16+8h..16+8h+7
  return f.v;
}

__device__ __forceinline__ v8f wmma_bf16(v16bf a, v16bf b, v8f c) {
  return __builtin_amdgcn_wmma_f32_16x16x32_bf16(false, a, false, b, (short)0, c, false, false);
}

// C/D layout: VGPR j -> lanes 0-15 (M=m0+j, N=n0+lane), lanes 16-31 (M=m0+8+j).
// Store into fragment-native bf16 LDS layout (element (m, n)).
__device__ __forceinline__ void store_tile_lds(uint32_t* Bout, int m0, int n0, v8f c,
                                               const float* bias, bool relu, int lane) {
  int n  = n0 + (lane & 15);
  int mh = (lane >> 4) << 3;
  uint16_t* o = (uint16_t*)Bout;
#pragma unroll
  for (int j = 0; j < 8; ++j) {
    int m = m0 + mh + j;
    float v = c[j];
    if (bias) v += bias[m];
    if (relu) v = fmaxf(v, 0.0f);
    o[((m >> 5) * KB + n * CS + ((m & 31) >> 1)) * 2 + (m & 1)] = f2bf(v);
  }
}

__device__ __forceinline__ void store_tile_f32(float* S, int m0, int n0, v8f c,
                                               float scale, int lane) {
  int n  = n0 + (lane & 15);
  int mh = (lane >> 4) << 3;
#pragma unroll
  for (int j = 0; j < 8; ++j) S[(m0 + mh + j) * 64 + n] = c[j] * scale;
}

__device__ __forceinline__ void gemm_2x4(const uint16_t* A, int lda, int m0,
                                         const uint32_t* B, int K, int lane, v8f acc[2][4]) {
  for (int k = 0; k < K; k += 32) {
    v16bf a0 = load_a(A, lda, m0,      k, lane);
    v16bf a1 = load_a(A, lda, m0 + 16, k, lane);
    v16bf b0 = load_b(B, k,  0, lane);
    v16bf b1 = load_b(B, k, 16, lane);
    v16bf b2 = load_b(B, k, 32, lane);
    v16bf b3 = load_b(B, k, 48, lane);
    acc[0][0] = wmma_bf16(a0, b0, acc[0][0]);
    acc[0][1] = wmma_bf16(a0, b1, acc[0][1]);
    acc[0][2] = wmma_bf16(a0, b2, acc[0][2]);
    acc[0][3] = wmma_bf16(a0, b3, acc[0][3]);
    acc[1][0] = wmma_bf16(a1, b0, acc[1][0]);
    acc[1][1] = wmma_bf16(a1, b1, acc[1][1]);
    acc[1][2] = wmma_bf16(a1, b2, acc[1][2]);
    acc[1][3] = wmma_bf16(a1, b3, acc[1][3]);
  }
}

__device__ __forceinline__ void gemm_1x4(const uint16_t* A, int lda, int m0,
                                         const uint32_t* B, int K, int lane, v8f acc[4]) {
  for (int k = 0; k < K; k += 32) {
    v16bf a0 = load_a(A, lda, m0, k, lane);
    v16bf b0 = load_b(B, k,  0, lane);
    v16bf b1 = load_b(B, k, 16, lane);
    v16bf b2 = load_b(B, k, 32, lane);
    v16bf b3 = load_b(B, k, 48, lane);
    acc[0] = wmma_bf16(a0, b0, acc[0]);
    acc[1] = wmma_bf16(a0, b1, acc[1]);
    acc[2] = wmma_bf16(a0, b2, acc[2]);
    acc[3] = wmma_bf16(a0, b3, acc[3]);
  }
}

// ---------------------------------------------------------------------------
// Prep: f32 -> bf16 weight conversion
// ---------------------------------------------------------------------------
__global__ void cvt_bf16(const float* __restrict__ src, uint16_t* __restrict__ dst, int n) {
  int i = blockIdx.x * blockDim.x + threadIdx.x;
  if (i < n) dst[i] = f2bf(src[i]);
}

// ---------------------------------------------------------------------------
// Prep: proxy path -> keyT[B,160,256] (bf16, zero-padded) and v[B,256,160]
// One block per (b, kp). Tiny (~0.4 GFLOP total), VALU is fine.
// ---------------------------------------------------------------------------
__global__ __launch_bounds__(256) void proxy_prep(
    const float* __restrict__ proxy,
    const float* __restrict__ w_fo1, const float* __restrict__ b_fo1,
    const float* __restrict__ w_fo2, const float* __restrict__ b_fo2,
    const float* __restrict__ w_fd,  const float* __restrict__ b_fd,
    uint16_t* __restrict__ keyT, uint16_t* __restrict__ vmat) {
  __shared__ float col[512];
  __shared__ float t1[256];
  int b  = blockIdx.x / 160;
  int kp = blockIdx.x % 160;
  int t  = threadIdx.x;
  uint16_t* kT = keyT + (size_t)b * 160 * 256;
  uint16_t* vm = vmat + (size_t)b * 256 * 160;
  if (kp >= 150) {                       // zero padding rows/cols (uniform branch)
    kT[kp * 256 + t] = 0;
    vm[t * 160 + kp] = 0;
    return;
  }
  col[t]       = proxy[((size_t)b * 512 + t)       * 150 + kp];
  col[t + 256] = proxy[((size_t)b * 512 + t + 256) * 150 + kp];
  __syncthreads();
  float s1 = b_fo1[t], sd = b_fd[t];
  for (int c = 0; c < 512; ++c) {
    float pv = col[c];
    s1 += w_fo1[t * 512 + c] * pv;
    sd += w_fd[t * 512 + c]  * pv;
  }
  t1[t] = fmaxf(s1, 0.0f);
  vm[t * 160 + kp] = f2bf(fmaxf(sd, 0.0f));
  __syncthreads();
  float s2 = b_fo2[t];
  for (int c = 0; c < 256; ++c) s2 += w_fo2[t * 256 + c] * t1[c];
  kT[kp * 256 + t] = f2bf(fmaxf(s2, 0.0f));
}

// ---------------------------------------------------------------------------
// Main fused kernel: one workgroup (256 thr = 8 waves) per 64-pixel tile.
// All intermediates live in 160 KB of LDS (2 workgroups / 320KB WGP);
// HBM traffic = X in + out only.
// ---------------------------------------------------------------------------
__global__ __launch_bounds__(256) void attn_main(
    const float* __restrict__ x,
    const uint16_t* __restrict__ w1b, const float* __restrict__ b1,
    const uint16_t* __restrict__ w2b, const float* __restrict__ b2,
    const uint16_t* __restrict__ keyT,   // [B][160][256]
    const uint16_t* __restrict__ vmat,   // [B][256][160]
    const uint16_t* __restrict__ wub, const float* __restrict__ bu,
    float* __restrict__ out) {
  // R0: X tile (16 K-blocks, 80KB) -> later sim f32 (160x64, 40KB) + P (25KB)
  // R1: q1 (8 K-blocks, 40KB) -> later ctx
  // R2: q2 (8 K-blocks, 40KB)
  __shared__ uint32_t R0[16 * KB];
  __shared__ uint32_t R1[8 * KB];
  __shared__ uint32_t R2[8 * KB];

  const int tid  = threadIdx.x;
  const int wave = tid >> 5;
  const int lane = tid & 31;
  const int b    = blockIdx.x >> 8;           // 256 tiles per batch (HW/64)
  const int p0   = (blockIdx.x & 255) << 6;

  // ---- Stage X tile: f32 global -> bf16 fragment-native LDS --------------
  for (int d = tid; d < 16384; d += 256) {
    int c2 = d >> 6, n = d & 63;              // K-pair index, pixel
    const float* xp = x + ((size_t)b * 512 + 2 * c2) * 16384 + p0 + n;
    R0[(c2 >> 4) * KB + n * CS + (c2 & 15)] = pack2bf(xp[0], xp[16384]);
  }
  __syncthreads();

  v8f acc[2][4];
  const int m0 = wave * 32;

  // ---- Phase 1: q1 = relu(W_fp1 @ X), M=256 K=512 ------------------------
#pragma unroll
  for (int i = 0; i < 2; ++i)
#pragma unroll
    for (int j = 0; j < 4; ++j) acc[i][j] = (v8f)0.0f;
  gemm_2x4(w1b, 512, m0, R0, 512, lane, acc);
#pragma unroll
  for (int i = 0; i < 2; ++i)
#pragma unroll
    for (int j = 0; j < 4; ++j)
      store_tile_lds(R1, m0 + 16 * i, 16 * j, acc[i][j], b1, true, lane);
  __syncthreads();

  // ---- Phase 2: q2 = relu(W_fp2 @ q1), M=256 K=256 -----------------------
#pragma unroll
  for (int i = 0; i < 2; ++i)
#pragma unroll
    for (int j = 0; j < 4; ++j) acc[i][j] = (v8f)0.0f;
  gemm_2x4(w2b, 256, m0, R1, 256, lane, acc);
#pragma unroll
  for (int i = 0; i < 2; ++i)
#pragma unroll
    for (int j = 0; j < 4; ++j)
      store_tile_lds(R2, m0 + 16 * i, 16 * j, acc[i][j], b2, true, lane);
  __syncthreads();

  // ---- Phase 3: sim = (keyT @ q2) / 16, M=160 K=256 (f32 into R0) --------
  float* sim = (float*)R0;
  const uint16_t* kT = keyT + (size_t)b * 160 * 256;
  for (int mt = wave; mt < 10; mt += 8) {
    v8f a4[4];
#pragma unroll
    for (int j = 0; j < 4; ++j) a4[j] = (v8f)0.0f;
    gemm_1x4(kT, 256, mt * 16, R2, 256, lane, a4);
#pragma unroll
    for (int j = 0; j < 4; ++j)
      store_tile_f32(sim, mt * 16, 16 * j, a4[j], 0.0625f, lane);   // Kc^-0.5 = 1/16
  }
  __syncthreads();

  // ---- Softmax over kp (masked to 150), P bf16 fragment-native -----------
  uint32_t* P = R0 + 10240;                   // after 40KB of sim (needs 6400 dw)
  if (tid < 64) {
    float mx = -1e30f;
    for (int kp = 0; kp < 150; ++kp) mx = fmaxf(mx, sim[kp * 64 + tid]);
    float s = 0.0f;
    for (int kp = 0; kp < 150; ++kp) s += __expf(sim[kp * 64 + tid] - mx);
    float inv = 1.0f / s;
    for (int kd = 0; kd < 80; ++kd) {         // K-pair index
      int k0 = 2 * kd;
      float e0 = (k0     < 150) ? __expf(sim[k0 * 64 + tid] - mx) * inv : 0.0f;
      float e1 = (k0 + 1 < 150) ? __expf(sim[(k0 + 1) * 64 + tid] - mx) * inv : 0.0f;
      P[(kd >> 4) * KB + tid * CS + (kd & 15)] = pack2bf(e0, e1);
    }
  }
  __syncthreads();

  // ---- Phase 4: ctx = v @ P, M=256 K=160 (no bias/relu), into R1 ---------
#pragma unroll
  for (int i = 0; i < 2; ++i)
#pragma unroll
    for (int j = 0; j < 4; ++j) acc[i][j] = (v8f)0.0f;
  gemm_2x4(vmat + (size_t)b * 256 * 160, 160, m0, P, 160, lane, acc);
#pragma unroll
  for (int i = 0; i < 2; ++i)
#pragma unroll
    for (int j = 0; j < 4; ++j)
      store_tile_lds(R1, m0 + 16 * i, 16 * j, acc[i][j], nullptr, false, lane);
  __syncthreads();

  // ---- Phase 5: out = relu(W_fu @ ctx + b_fu), M=512 K=256 ---------------
  for (int half = 0; half < 2; ++half) {
    int m5 = wave * 64 + half * 32;
#pragma unroll
    for (int i = 0; i < 2; ++i)
#pragma unroll
      for (int j = 0; j < 4; ++j) acc[i][j] = (v8f)0.0f;
    gemm_2x4(wub, 256, m5, R1, 256, lane, acc);
#pragma unroll
    for (int i = 0; i < 2; ++i)
#pragma unroll
      for (int j = 0; j < 4; ++j) {
        int n  = 16 * j + (lane & 15);
        int mh = (lane >> 4) << 3;
#pragma unroll
        for (int r = 0; r < 8; ++r) {
          int m = m5 + 16 * i + mh + r;
          float v = fmaxf(acc[i][j][r] + bu[m], 0.0f);
          out[((size_t)b * 512 + m) * 16384 + p0 + n] = v;
        }
      }
  }
}

// ---------------------------------------------------------------------------
extern "C" void kernel_launch(void* const* d_in, const int* in_sizes, int n_in,
                              void* d_out, int out_size, void* d_ws, size_t ws_size,
                              hipStream_t stream) {
  const float* x     = (const float*)d_in[0];
  const float* proxy = (const float*)d_in[1];
  const float* w_fp1 = (const float*)d_in[2];
  const float* b_fp1 = (const float*)d_in[3];
  const float* w_fp2 = (const float*)d_in[4];
  const float* b_fp2 = (const float*)d_in[5];
  const float* w_fo1 = (const float*)d_in[6];
  const float* b_fo1 = (const float*)d_in[7];
  const float* w_fo2 = (const float*)d_in[8];
  const float* b_fo2 = (const float*)d_in[9];
  const float* w_fd  = (const float*)d_in[10];
  const float* b_fd  = (const float*)d_in[11];
  const float* w_fu  = (const float*)d_in[12];
  const float* b_fu  = (const float*)d_in[13];
  float* out = (float*)d_out;

  uint16_t* w1b  = (uint16_t*)d_ws;                 // 256*512
  uint16_t* w2b  = w1b + 131072;                    // 256*256
  uint16_t* wub  = w2b + 65536;                     // 512*256
  uint16_t* keyT = wub + 131072;                    // 8*160*256
  uint16_t* vmat = keyT + 8 * 160 * 256;            // 8*256*160

  cvt_bf16<<<131072 / 256, 256, 0, stream>>>(w_fp1, w1b, 131072);
  cvt_bf16<<<65536 / 256, 256, 0, stream>>>(w_fp2, w2b, 65536);
  cvt_bf16<<<131072 / 256, 256, 0, stream>>>(w_fu, wub, 131072);
  proxy_prep<<<8 * 160, 256, 0, stream>>>(proxy, w_fo1, b_fo1, w_fo2, b_fo2,
                                          w_fd, b_fd, keyT, vmat);
  attn_main<<<2048, 256, 0, stream>>>(x, w1b, b_fp1, w2b, b_fp2, keyT, vmat,
                                      wub, b_fu, out);
}
